// ScaledDotProductAttention_721554505830
// MI455X (gfx1250) — matrix operand order repaired and checked by reading, not compiled, not run on hardware
//
#include <hip/hip_runtime.h>
#include <hip/hip_bf16.h>

typedef __attribute__((ext_vector_type(16))) _Float16 v16h;
typedef __attribute__((ext_vector_type(8)))  _Float16 v8h;
typedef __attribute__((ext_vector_type(4)))  _Float16 v4h;
typedef __attribute__((ext_vector_type(8)))  float    v8f;
typedef __attribute__((ext_vector_type(4)))  float    v4f;

#define BH_   256
#define T_    1024
#define D_    128
#define SCALE_ 0.08838834764831843f   // 1/sqrt(128)
#define MASKN_ (-4294967296.0f)       // float32(-2^32+1)

#define KS_ 136   // Kbuf row stride (halves): 272B = 68 banks -> 4n mod 64, conflict-free
#define VS_ 40    // Vt  row stride (halves): 80B  = 20 banks
#define PS_ 40    // Pbuf row stride (halves)

static __device__ __forceinline__ v16h lds_b16(const _Float16* p) {
    // two 8-half runs (B-matrix fragment: lane n=l&15 holds col n, halves 0..7 -> K=g*8+i, halves 8..15 -> K=16+g*8+i)
    v8h a = *(const v8h*)p;
    v8h b = *(const v8h*)(p + 16);
    return __builtin_shufflevector(a, b, 0,1,2,3,4,5,6,7,8,9,10,11,12,13,14,15);
}

static __device__ __forceinline__ v16h load_a_f32(const float* row, int g, int dbase) {
    // A-matrix fragment from a global f32 row: halves 0..7 -> K=g*8+i, halves 8..15 -> K=16+g*8+i
    v4f a0 = *(const v4f*)(row + dbase + g * 8);
    v4f a1 = *(const v4f*)(row + dbase + g * 8 + 4);
    v4f b0 = *(const v4f*)(row + dbase + 16 + g * 8);
    v4f b1 = *(const v4f*)(row + dbase + 16 + g * 8 + 4);
    v16h r;
#pragma unroll
    for (int i = 0; i < 4; ++i) {
        r[i]      = (_Float16)a0[i];
        r[4 + i]  = (_Float16)a1[i];
        r[8 + i]  = (_Float16)b0[i];
        r[12 + i] = (_Float16)b1[i];
    }
    return r;
}

static __device__ __forceinline__ float rmax16(float x) {
#pragma unroll
    for (int m = 1; m < 16; m <<= 1) x = fmaxf(x, __shfl_xor(x, m, 32));
    return x;
}
static __device__ __forceinline__ float rsum16(float x) {
#pragma unroll
    for (int m = 1; m < 16; m <<= 1) x += __shfl_xor(x, m, 32);
    return x;
}

__launch_bounds__(128)
__global__ void ScaledDotProductAttention_721554505830_kernel(
    const float* __restrict__ Q, const float* __restrict__ K,
    const float* __restrict__ V, const int* __restrict__ masks,
    float* __restrict__ out)
{
    __shared__ __align__(16) _Float16 Kbuf[32 * KS_];       // K chunk, row-major [k][d], f16
    __shared__ __align__(16) _Float16 Vt[D_ * VS_];         // V chunk, transposed [d][k], f16
    __shared__ __align__(16) _Float16 Pbuf[4 * 16 * PS_];   // per-wave P tile [row][k], f16

    const int t    = threadIdx.x;         // 0..127
    const int lane = t & 31;
    const int w    = t >> 5;              // wave id 0..3
    const int n    = lane & 15;           // N / column index inside 16-wide tile
    const int g    = lane >> 4;           // half-wave id
    const int qtile = blockIdx.x;         // 0..15   (64 q rows each)
    const int bh    = blockIdx.y;         // 0..255
    const int q0w   = qtile * 64 + w * 16;

    const int* maskrow = masks + (size_t)(bh & 31) * T_;

    // ---- load Q fragments (A layout, f32 -> f16), 16 rows x 128 d per wave ----
    const float* Qrow = Q + ((size_t)bh * T_ + q0w + n) * D_;   // row m = lane&15
    v16h qfrag[4];
#pragma unroll
    for (int dc = 0; dc < 4; ++dc) qfrag[dc] = load_a_f32(Qrow, g, dc * 32);

    v8f ofrag[8];
#pragma unroll
    for (int j = 0; j < 8; ++j) ofrag[j] = (v8f){};
    float m_run[8], l_run[8];
#pragma unroll
    for (int r = 0; r < 8; ++r) { m_run[r] = -INFINITY; l_run[r] = 0.0f; }

    const int kmax = (qtile + 1) * 64 < T_ ? (qtile + 1) * 64 : T_;  // causal upper bound

    for (int kc = 0; kc < kmax; kc += 32) {
        __syncthreads();   // previous iteration's LDS reads complete before restaging

        // ---- cooperative staging: K chunk (row-major) and V chunk (transposed) ----
        const float* Kg = K + ((size_t)bh * T_ + kc) * D_;
        const float* Vg = V + ((size_t)bh * T_ + kc) * D_;
#pragma unroll
        for (int rep = 0; rep < 8; ++rep) {
            int e  = rep * 512 + t * 4;     // 0..4095 over 32x128 elements
            int kk = e >> 7, dd = e & 127;
            v4f kv = *(const v4f*)(Kg + (size_t)kk * D_ + dd);
            v4h kh;
#pragma unroll
            for (int i = 0; i < 4; ++i) kh[i] = (_Float16)kv[i];
            *(v4h*)(Kbuf + kk * KS_ + dd) = kh;

            v4f vv = *(const v4f*)(Vg + (size_t)kk * D_ + dd);
#pragma unroll
            for (int i = 0; i < 4; ++i) Vt[(dd + i) * VS_ + kk] = (_Float16)vv[i];
        }
        __syncthreads();

        // ---- scores: two 16x16 tiles (k cols kc..kc+31), K-dim = D = 4 x 32 ----
        v8f s[2];
#pragma unroll
        for (int t16 = 0; t16 < 2; ++t16) {
            v8f c = (v8f){};
            const _Float16* Kb = Kbuf + (t16 * 16 + n) * KS_ + g * 8;
#pragma unroll
            for (int dc = 0; dc < 4; ++dc) {
                v16h b = lds_b16(Kb + dc * 32);
                c = __builtin_amdgcn_wmma_f32_16x16x32_f16(false, qfrag[dc],
                                                           false, b,
                                                           (short)0, c, false, false);
            }
            const int col = kc + t16 * 16 + n;
            const float km = MASKN_ * (float)maskrow[col];
#pragma unroll
            for (int r = 0; r < 8; ++r) {
                float sv = c[r] * SCALE_ + km;
                int qr = q0w + r + 8 * g;
                s[t16][r] = (col > qr) ? MASKN_ : sv;
            }
        }

        // ---- online softmax update + P write ----
        float alpha[8];
#pragma unroll
        for (int r = 0; r < 8; ++r) {
            float rm   = rmax16(fmaxf(s[0][r], s[1][r]));
            float mnew = fmaxf(m_run[r], rm);
            float a    = __expf(m_run[r] - mnew);
            float p0   = __expf(s[0][r] - mnew);
            float p1   = __expf(s[1][r] - mnew);
            float rs   = rsum16(p0 + p1);
            l_run[r]   = l_run[r] * a + rs;
            m_run[r]   = mnew;
            alpha[r]   = a;
            _Float16* pw = Pbuf + (size_t)(w * 16 + r + 8 * g) * PS_;
            pw[n]      = (_Float16)p0;
            pw[16 + n] = (_Float16)p1;
        }
#pragma unroll
        for (int j = 0; j < 8; ++j)
#pragma unroll
            for (int r = 0; r < 8; ++r) ofrag[j][r] *= alpha[r];

        __syncthreads();   // Pbuf visible; also orders before Vt reads

        // ---- O += P (16x32, A layout) x V (32x16 per d-tile, B layout from Vt) ----
        const _Float16* pa = Pbuf + (size_t)(w * 16 + n) * PS_;
        v8h plo = *(const v8h*)(pa + g * 8);
        v8h phi = *(const v8h*)(pa + 16 + g * 8);
        v16h pf = __builtin_shufflevector(plo, phi, 0,1,2,3,4,5,6,7,8,9,10,11,12,13,14,15);
#pragma unroll
        for (int j = 0; j < 8; ++j) {
            v16h bv = lds_b16(Vt + (j * 16 + n) * VS_ + g * 8);
            ofrag[j] = __builtin_amdgcn_wmma_f32_16x16x32_f16(false, pf, false, bv,
                                                              (short)0, ofrag[j], false, false);
        }
    }

    // ---- epilogue: denominator correction for the untouched all-MASKN_ region,
    //      then normalize and store (C layout: row r+8g, col j*16+n) ----
    const float lrem = (float)(T_ - kmax);
#pragma unroll
    for (int r = 0; r < 8; ++r) {
        float l   = l_run[r] + lrem * __expf(MASKN_ - m_run[r]);
        float inv = 1.0f / l;
        int qr = q0w + r + 8 * g;
        float* orow = out + ((size_t)bh * T_ + qr) * D_ + n;
#pragma unroll
        for (int j = 0; j < 8; ++j) orow[j * 16] = ofrag[j][r] * inv;
    }
}

extern "C" void kernel_launch(void* const* d_in, const int* in_sizes, int n_in,
                              void* d_out, int out_size, void* d_ws, size_t ws_size,
                              hipStream_t stream) {
    (void)in_sizes; (void)n_in; (void)out_size; (void)d_ws; (void)ws_size;
    const float* q     = (const float*)d_in[0];
    const float* k     = (const float*)d_in[1];
    const float* v     = (const float*)d_in[2];
    const int*   masks = (const int*)d_in[3];
    float* out = (float*)d_out;

    dim3 grid(T_ / 64, BH_);   // 16 q-tiles x 256 batch-heads
    dim3 block(128);           // 4 waves; each wave owns a 16-row q tile
    ScaledDotProductAttention_721554505830_kernel<<<grid, block, 0, stream>>>(q, k, v, masks, out);
}
